// MultiHeadAttention_21096879358514
// MI455X (gfx1250) — matrix operand order repaired and checked by reading, not compiled, not run on hardware
//
#include <hip/hip_runtime.h>

// MHA for MI455X (gfx1250): bf16 WMMA everywhere, fused attention with the
// 32x2048 f32 score slab resident in LDS (256KB of the 320KB/WGP budget) so
// the mandatory 1.07GB attn tensor touches HBM exactly once (as b128 stores).
//
// Round 3: staging rewritten for batched vector loads + constant-offset
// addressing (no per-element 64-bit address math), B tiles staged straight
// into WMMA fragment order with one ds_store_b128 per thread, mask applied
// in the softmax pass from packed u32 reads, vectorized softmax/attn-write.

typedef __attribute__((ext_vector_type(16))) __bf16 v16bf;
typedef __attribute__((ext_vector_type(8)))  __bf16 v8bf;
typedef __attribute__((ext_vector_type(4)))  __bf16 v4bf;
typedef __attribute__((ext_vector_type(8)))  float  v8f;
typedef __attribute__((ext_vector_type(4)))  float  v4f;

#define NHEAD   16
#define HDIM    64
#define DMODEL  1024
#define SEQ     2048
#define BATCH   4
#define ATTN_SCALE 0.125f   // 1/sqrt(64)

static __device__ inline v8f wmma_bf16(v16bf a, v16bf b, v8f c) {
    return __builtin_amdgcn_wmma_f32_16x16x32_bf16(
        /*neg_a=*/false, a, /*neg_b=*/false, b,
        /*c_mod=*/(short)0, c, /*reuse_a=*/false, /*reuse_b=*/false);
}

// ---- fragment loaders (ISA 7.12.2 layouts, wave32) ----

// A matrix 16x32 bf16: lanes 0-15 row M=lane K={0..7,16..23}; lanes 16-31
// row M=lane-16 K={8..15,24..31}. Two contiguous 16B reads per lane.
static __device__ inline v16bf frag_a_bf16(const __bf16* p, int row0, int ld, int col0) {
    const int lane = threadIdx.x & 31;
    const int m    = lane & 15;
    const int kb   = (lane < 16) ? 0 : 8;
    const __bf16* r = p + (size_t)(row0 + m) * ld + col0 + kb;
    v16bf a;
#pragma unroll
    for (int i = 0; i < 8; ++i) a[i] = r[i];
#pragma unroll
    for (int i = 0; i < 8; ++i) a[8 + i] = r[16 + i];
    return a;
}

// A fragment from a lane-resolved f32 pointer (LDS score slab), cvt to bf16.
static __device__ inline v16bf frag_a_f32_contig(const float* r) {
    v16bf a;
#pragma unroll
    for (int i = 0; i < 8; ++i) a[i] = (__bf16)r[i];
#pragma unroll
    for (int i = 0; i < 8; ++i) a[8 + i] = (__bf16)r[16 + i];
    return a;
}

// B fragment from a lane-resolved pointer: bf16 r[0..7] and r[16..23] per lane.
static __device__ inline v16bf frag_b_contig(const __bf16* r) {
    v16bf b;
#pragma unroll
    for (int i = 0; i < 16; ++i) b[i] = r[(i < 8) ? i : i + 8];
    return b;
}

// ------------------------------------------------------------------
// GEMM: out[M,N] = A_f32[M,K] @ W_f32[K,N] + bias.
// out_mode: 0 = bf16 row-major, 1 = f32 row-major, 2 = bf16 V-transposed
// ([B,H,HD,S], hardwired to M=8192,N=1024 shapes).
// Block tile 128x64, k-step 32, 256 threads = 8 waves.
// ------------------------------------------------------------------
#define GT_M 128
#define GT_N 64
#define GT_K 32
#define LDA_S 40   // bf16 elems; 80B rows keep alignment for frag reads

__global__ __launch_bounds__(256) void gemm_bias_kernel(
    const float* __restrict__ A, const float* __restrict__ W,
    const float* __restrict__ bias, __bf16* __restrict__ outb,
    float* __restrict__ outf, int M, int N, int K, int out_mode)
{
    __shared__ __bf16 sA[GT_M * LDA_S];
    __shared__ __bf16 sBp[4 * 512];   // 4 n-tiles, fragment-packed, 4KB

    const int tid  = threadIdx.x;
    const int wave = tid >> 5;
    const int lane = tid & 31;
    const int m0   = blockIdx.y * GT_M;
    const int n0   = blockIdx.x * GT_N;

    // k-invariant staging addresses, advanced by constant strides per step.
    // A: thread owns rows (tid>>3)+32i, cols (tid&7)*4..+3  (16B vector loads)
    const int ar = tid >> 3, ac = (tid & 7) << 2;
    const float* aG = A + (size_t)(m0 + ar) * K + ac;
    __bf16*      aL = sA + ar * LDA_S + ac;
    // B: thread owns n = tid&63, k = (tid>>6)*8..+7 (8-deep column, coalesced
    // across lanes), stored as ONE b128 directly in fragment order.
    const int bn = tid & 63, bk0 = (tid >> 6) << 3;
    const float* wG = W + (size_t)bk0 * N + n0 + bn;
    __bf16*      bL = sBp + (bn >> 4) * 512 +
                      ((((bk0 & 16) >> 4) * 32 + (bn & 15) + (bk0 & 8) * 2) * 8);

    v8f acc[4] = {};

    for (int kc = 0; kc < K; kc += GT_K) {
        __syncthreads();
        // Gather first (independent loads in flight), then convert+store.
        v4f av[4];
#pragma unroll
        for (int i = 0; i < 4; ++i)
            av[i] = *(const v4f*)(aG + (size_t)(i * 32) * K);
        float wv[8];
#pragma unroll
        for (int i = 0; i < 8; ++i)
            wv[i] = wG[(size_t)i * N];

#pragma unroll
        for (int i = 0; i < 4; ++i) {
            v4bf t;
#pragma unroll
            for (int e = 0; e < 4; ++e) t[e] = (__bf16)av[i][e];
            *(v4bf*)(aL + i * 32 * LDA_S) = t;
        }
        {
            v8bf t;
#pragma unroll
            for (int e = 0; e < 8; ++e) t[e] = (__bf16)wv[e];
            *(v8bf*)bL = t;                       // one ds_store_b128
        }
        // Prefetch next k-step tiles (global_prefetch_b8).
        if (kc + GT_K < K) {
            __builtin_prefetch(aG + GT_K, 0, 1);
            __builtin_prefetch(wG + (size_t)GT_K * N, 0, 1);
        }
        aG += GT_K;
        wG += (size_t)GT_K * N;
        __syncthreads();

        v16bf a = frag_a_bf16(sA, wave * 16, LDA_S, 0);
#pragma unroll
        for (int j = 0; j < 4; ++j) {
            const __bf16* tile = sBp + j * 512;
            v16bf b = frag_b_contig(tile + lane * 8);        // chunk 0
            // chunk 1 lives at +256; frag_b_contig only covers 16 contiguous,
            // so assemble explicitly:
            v16bf bb;
#pragma unroll
            for (int e = 0; e < 8; ++e) bb[e] = b[e];
#pragma unroll
            for (int e = 0; e < 8; ++e) bb[8 + e] = tile[256 + lane * 8 + e];
            acc[j] = wmma_bf16(a, bb, acc[j]);
        }
    }

    // C/D layout: VGPR r -> row r (lanes 0-15) / row r+8 (lanes 16-31).
    const int rowhi = (lane < 16) ? 0 : 8;
    const int col   = lane & 15;
#pragma unroll
    for (int j = 0; j < 4; ++j) {
        const int nn = n0 + j * 16 + col;
        const float bv = bias[nn];
#pragma unroll
        for (int r = 0; r < 8; ++r) {
            const int mm = m0 + wave * 16 + r + rowhi;
            const float v = acc[j][r] + bv;
            if (out_mode == 1) {
                outf[(size_t)mm * N + nn] = v;
            } else if (out_mode == 2) {
                // Vt[((b*NHEAD+h)*HDIM + hd) * SEQ + s]
                const int bb = mm >> 11, s  = mm & (SEQ - 1);
                const int hh = nn >> 6,  hd = nn & (HDIM - 1);
                outb[(((size_t)bb * NHEAD + hh) * HDIM + hd) * SEQ + s] = (__bf16)v;
            } else {
                outb[(size_t)mm * N + nn] = (__bf16)v;
            }
        }
    }
}

// ------------------------------------------------------------------
// Fused attention: one workgroup = (batch b, head h, 32 query rows).
// Phase 1: raw scores = Q.K^T -> 32x2048 f32 slab in LDS.
// Phase 2: scale+mask folded into softmax max-pass (mask read as packed
//          u32), vectorized; single b128 HBM write of attn.
// Phase 3: context = attn(LDS->bf16) @ Vt via WMMA -> f32 ctx buffer.
// ------------------------------------------------------------------
__global__ __launch_bounds__(256) void attn_kernel(
    const __bf16* __restrict__ Qb, const __bf16* __restrict__ Kb,
    const __bf16* __restrict__ Vt, const unsigned char* __restrict__ mask,
    float* __restrict__ attn_out, float* __restrict__ ctx)
{
    extern __shared__ float sc[];          // 32 * SEQ floats = 256 KB
    const int tid  = threadIdx.x;
    const int wave = tid >> 5;
    const int lane = tid & 31;
    const int qb   = blockIdx.x * 32;      // first query row of this block
    const int h    = blockIdx.y;
    const int b    = blockIdx.z;
    const int hd0  = h * HDIM;
    const int rowb = b * SEQ;              // row base into Qb/Kb

    const int rowhi = (lane < 16) ? 0 : 8;
    const int col   = lane & 15;
    const int n     = lane & 15;
    const int kb    = (lane < 16) ? 0 : 8;

    // ---------- Phase 1: raw scores ----------
    v16bf qa[2][2];
#pragma unroll
    for (int mt = 0; mt < 2; ++mt)
#pragma unroll
        for (int kc = 0; kc < 2; ++kc)
            qa[mt][kc] = frag_a_bf16(Qb, rowb + qb + mt * 16, DMODEL, hd0 + kc * 32);

    // Lane-resolved K^T base: B[k][nn] = Kb[rowb + n0 + nn][hd0 + k].
    const __bf16* kbase = Kb + (size_t)(rowb + wave * 256 + n) * DMODEL + hd0 + kb;

    for (int nt = 0; nt < 16; ++nt) {
        const int n0 = wave * 256 + nt * 16;        // key-column base in [0,SEQ)
        v8f a0 = {}, a1 = {};
#pragma unroll
        for (int kc = 0; kc < 2; ++kc) {
            v16bf bk = frag_b_contig(kbase + nt * 16 * DMODEL + kc * 32);
            a0 = wmma_bf16(qa[0][kc], bk, a0);
            a1 = wmma_bf16(qa[1][kc], bk, a1);
        }
        const int c = n0 + col;
#pragma unroll
        for (int r = 0; r < 8; ++r) {
            sc[(r + rowhi) * SEQ + c]      = a0[r];
            sc[(16 + r + rowhi) * SEQ + c] = a1[r];
        }
    }
    __syncthreads();

    // ---------- Phase 2: scale+mask+softmax (wave w: rows 4w..4w+3) -------
    for (int rr = 0; rr < 4; ++rr) {
        const int row = wave * 4 + rr;
        v4f* srow4 = (v4f*)(sc + (size_t)row * SEQ);
        const unsigned int* mrow4 =
            (const unsigned int*)(mask + ((size_t)b * SEQ + qb + row) * SEQ);
        float mx = -1e30f;
        for (int j = lane; j < SEQ / 4; j += 32) {
            v4f v = srow4[j];
            const unsigned int mu = mrow4[j];
#pragma unroll
            for (int e = 0; e < 4; ++e) {
                const float s = ((mu >> (8 * e)) & 255u) ? -1e9f : v[e] * ATTN_SCALE;
                v[e] = s;
                mx = fmaxf(mx, s);
            }
            srow4[j] = v;
        }
#pragma unroll
        for (int o = 16; o > 0; o >>= 1) mx = fmaxf(mx, __shfl_xor(mx, o));
        float sum = 0.0f;
        for (int j = lane; j < SEQ / 4; j += 32) {
            v4f v = srow4[j];
#pragma unroll
            for (int e = 0; e < 4; ++e) {
                const float ee = __expf(v[e] - mx);
                v[e] = ee;
                sum += ee;
            }
            srow4[j] = v;
        }
#pragma unroll
        for (int o = 16; o > 0; o >>= 1) sum += __shfl_xor(sum, o);
        const float inv = 1.0f / sum;
        v4f* arow4 = (v4f*)(attn_out +
                            (((size_t)b * NHEAD + h) * SEQ + (qb + row)) * SEQ);
        for (int j = lane; j < SEQ / 4; j += 32) {
            v4f v = srow4[j];
#pragma unroll
            for (int e = 0; e < 4; ++e) v[e] *= inv;
            srow4[j] = v;         // normalized prob kept in LDS for phase 3
            arow4[j] = v;         // the one mandatory HBM write (b128)
        }
    }
    __syncthreads();

    // ---------- Phase 3: context = attn @ V ----------
    // 8 output 16x16 tiles (2 m x 4 n), one per wave. K = 2048 in 64 chunks.
    const int mt  = wave & 1;
    const int nt4 = wave >> 1;
    const int hdrow0 = (b * NHEAD + h) * HDIM + nt4 * 16;
    const __bf16* vbase = Vt + (size_t)(hdrow0 + n) * SEQ + kb;
    const float*  abase = sc + (size_t)(mt * 16 + (lane & 15)) * SEQ +
                          ((lane < 16) ? 0 : 8);
    v8f acc = {};
    for (int kc = 0; kc < SEQ; kc += 32) {
        v16bf a = frag_a_f32_contig(abase + kc);
        v16bf v = frag_b_contig(vbase + kc);
        acc = wmma_bf16(a, v, acc);
    }
#pragma unroll
    for (int r = 0; r < 8; ++r) {
        const int mm = rowb + qb + mt * 16 + r + rowhi;
        const int nn = hd0 + nt4 * 16 + col;
        ctx[(size_t)mm * DMODEL + nn] = acc[r];
    }
}

// ------------------------------------------------------------------
extern "C" void kernel_launch(void* const* d_in, const int* in_sizes, int n_in,
                              void* d_out, int out_size, void* d_ws, size_t ws_size,
                              hipStream_t stream)
{
    const float* query = (const float*)d_in[0];
    const float* key   = (const float*)d_in[1];
    const float* value = (const float*)d_in[2];
    const unsigned char* mask = (const unsigned char*)d_in[3];
    const float* Wq = (const float*)d_in[4];
    const float* bq = (const float*)d_in[5];
    const float* Wk = (const float*)d_in[6];
    const float* bk = (const float*)d_in[7];
    const float* Wv = (const float*)d_in[8];
    const float* bv = (const float*)d_in[9];
    const float* Wo = (const float*)d_in[10];
    const float* bo = (const float*)d_in[11];

    float* outputs = (float*)d_out;                               // [B,S,D]
    float* attn    = outputs + (size_t)BATCH * SEQ * DMODEL;      // [B,H,S,S]

    const int M = BATCH * SEQ;                                    // 8192
    // Workspace: Qb/Kb bf16 row-major, Vt bf16 transposed, ctx f32 = 80MB.
    __bf16* Qb = (__bf16*)d_ws;
    __bf16* Kb = Qb + (size_t)M * DMODEL;
    __bf16* Vt = Kb + (size_t)M * DMODEL;
    float*  ctx = (float*)(Vt + (size_t)M * DMODEL);

    dim3 blk(256);
    dim3 ggrid(DMODEL / GT_N, M / GT_M);                          // (16, 64)

    gemm_bias_kernel<<<ggrid, blk, 0, stream>>>(query, Wq, bq, Qb, nullptr,
                                                M, DMODEL, DMODEL, 0);
    gemm_bias_kernel<<<ggrid, blk, 0, stream>>>(key,   Wk, bk, Kb, nullptr,
                                                M, DMODEL, DMODEL, 0);
    gemm_bias_kernel<<<ggrid, blk, 0, stream>>>(value, Wv, bv, Vt, nullptr,
                                                M, DMODEL, DMODEL, 2);

    dim3 agrid(SEQ / 32, NHEAD, BATCH);                           // (64,16,4)
    size_t lds = (size_t)32 * SEQ * sizeof(float);                // 256 KB
    attn_kernel<<<agrid, blk, lds, stream>>>(Qb, Kb, Vt, mask, attn, ctx);

    gemm_bias_kernel<<<ggrid, blk, 0, stream>>>(ctx, Wo, bo, nullptr, outputs,
                                                M, DMODEL, DMODEL, 1);
}
